// KEP_GAT_PNA_CE_27487790694539
// MI455X (gfx1250) — matrix operand order repaired and checked by reading, not compiled, not run on hardware
//
#include <hip/hip_runtime.h>

// ===========================================================================
// KEP GAT/PNA edge-scoring pipeline for MI455X (gfx1250, wave32, WMMA).
//
// Dense node transforms (GAT Wl/Wr, FCN) use v_wmma_f32_16x16x32_f16 with
// K templated (16 zero-padded / 32), A-fragments via b128 global loads,
// B-fragments via LDS-staged weights (ds_load). Edge gather/scatter phases
// are memory-bound VALU kernels with vectorized b128/b64 accesses and float
// atomics (add f32; ordered min/max via the sign-aware int/uint trick).
// ===========================================================================

typedef __attribute__((ext_vector_type(16))) _Float16 v16h;
typedef __attribute__((ext_vector_type(8)))  float    v8f;

#define DEVINL __device__ __forceinline__

DEVINL void atomicMaxF(float* a, float v) {
  if (v >= 0.f) atomicMax((int*)a, __float_as_int(v));
  else          atomicMin((unsigned int*)a, __float_as_uint(v));
}
DEVINL void atomicMinF(float* a, float v) {
  if (v >= 0.f) atomicMin((int*)a, __float_as_int(v));
  else          atomicMax((unsigned int*)a, __float_as_uint(v));
}
DEVINL float4 ld4(const float* p) { return *(const float4*)p; }

// ---------------------------------------------------------------------------
__global__ void fill_f32(float* __restrict__ p, float v, int n) {
  int i = blockIdx.x * blockDim.x + threadIdx.x;
  if (i < n) p[i] = v;
}

__global__ void avg_log_kernel(const int* __restrict__ deg, int bins,
                               float* __restrict__ out) {
  if (blockIdx.x == 0 && threadIdx.x == 0) {
    float num = 0.f, den = 0.f;
    for (int b = 0; b < bins; ++b) {
      float f = (float)deg[b];
      num += log1pf((float)b) * f;
      den += f;
    }
    out[0] = num / den;
  }
}

// ---------------------------------------------------------------------------
// PNA edge pass: both directions in one sweep over edges.
__global__ void pna_edge_kernel(
    const float* __restrict__ x, const float* __restrict__ ew,
    const int* __restrict__ ei, int E_,
    const float* __restrict__ We0, const float* __restrict__ be0,
    const float* __restrict__ Wpre0, const float* __restrict__ bpre0,
    const float* __restrict__ We1, const float* __restrict__ be1,
    const float* __restrict__ Wpre1, const float* __restrict__ bpre1,
    float* cnt0, float* sum0, float* sq0, float* mn0, float* mx0,
    float* cnt1, float* sum1, float* sq1, float* mn1, float* mx1) {
  int e = blockIdx.x * blockDim.x + threadIdx.x;
  if (e >= E_) return;
  int s = ei[e], d = ei[E_ + e];
  float w = ew[e];
  float xs[5], xd[5];
#pragma unroll
  for (int c = 0; c < 5; ++c) { xs[c] = x[s * 5 + c]; xd[c] = x[d * 5 + c]; }

  {  // direction 0 (pna): msg [x_dst, x_src, enc0] -> dst
    float enc[5];
#pragma unroll
    for (int j = 0; j < 5; ++j) enc[j] = w * We0[j] + be0[j];
#pragma unroll
    for (int j = 0; j < 5; ++j) {
      float m = bpre0[j];
#pragma unroll
      for (int i = 0; i < 5; ++i)
        m += xd[i] * Wpre0[i * 5 + j] + xs[i] * Wpre0[(5 + i) * 5 + j] +
             enc[i] * Wpre0[(10 + i) * 5 + j];
      atomicAdd(&sum0[d * 5 + j], m);
      atomicAdd(&sq0[d * 5 + j], m * m);
      atomicMinF(&mn0[d * 5 + j], m);
      atomicMaxF(&mx0[d * 5 + j], m);
    }
    atomicAdd(&cnt0[d], 1.f);
  }
  {  // direction 1 (pna_ce): msg [x_src, x_dst, enc1] -> src
    float enc[5];
#pragma unroll
    for (int j = 0; j < 5; ++j) enc[j] = w * We1[j] + be1[j];
#pragma unroll
    for (int j = 0; j < 5; ++j) {
      float m = bpre1[j];
#pragma unroll
      for (int i = 0; i < 5; ++i)
        m += xs[i] * Wpre1[i * 5 + j] + xd[i] * Wpre1[(5 + i) * 5 + j] +
             enc[i] * Wpre1[(10 + i) * 5 + j];
      atomicAdd(&sum1[s * 5 + j], m);
      atomicAdd(&sq1[s * 5 + j], m * m);
      atomicMinF(&mn1[s * 5 + j], m);
      atomicMaxF(&mx1[s * 5 + j], m);
    }
    atomicAdd(&cnt1[s], 1.f);
  }
}

// PNA node pass: scalers -> Wpost(65x8) -> Wlin(8x8) -> relu into h1 columns
__global__ void pna_node_kernel(
    const float* __restrict__ x, int N_,
    const float* __restrict__ cnt, const float* __restrict__ sum,
    const float* __restrict__ sq, const float* __restrict__ mn,
    const float* __restrict__ mx,
    const float* __restrict__ Wpost, const float* __restrict__ bpost,
    const float* __restrict__ Wlin, const float* __restrict__ blin,
    const float* __restrict__ avg_log, float* __restrict__ Y,
    int ldy, int ycol) {
  int i = blockIdx.x * blockDim.x + threadIdx.x;
  if (i >= N_) return;
  float c = cnt[i];
  float deg = fmaxf(c, 1.f);
  bool has = c > 0.f;
  float in65[65];
#pragma unroll
  for (int j = 0; j < 5; ++j) in65[j] = x[i * 5 + j];
  float agg[20];
#pragma unroll
  for (int j = 0; j < 5; ++j) {
    float mean = sum[i * 5 + j] / deg;
    float mean2 = sq[i * 5 + j] / deg;
    float var = fmaxf(mean2 - mean * mean, 0.f);
    agg[j] = mean;
    agg[5 + j] = has ? mn[i * 5 + j] : 0.f;
    agg[10 + j] = has ? mx[i * 5 + j] : 0.f;
    agg[15 + j] = sqrtf(var + 1e-5f);
  }
  float al = avg_log[0];
  float lg = log1pf(deg);
  float amp = lg / al, att = al / lg;
#pragma unroll
  for (int k = 0; k < 20; ++k) {
    in65[5 + k] = agg[k];
    in65[25 + k] = agg[k] * amp;
    in65[45 + k] = agg[k] * att;
  }
  float t8[8];
#pragma unroll
  for (int j = 0; j < 8; ++j) {
    float a = bpost[j];
    for (int k = 0; k < 65; ++k) a += in65[k] * Wpost[k * 8 + j];
    t8[j] = a;
  }
  float o8[8];
#pragma unroll
  for (int j = 0; j < 8; ++j) {
    float a = blin[j];
#pragma unroll
    for (int k = 0; k < 8; ++k) a += t8[k] * Wlin[k * 8 + j];
    o8[j] = fmaxf(a, 0.f);
  }
  float* yrow = Y + i * ldy + ycol;  // 32B-aligned (ycol 0 or 8)
  *(float4*)(yrow + 0) = make_float4(o8[0], o8[1], o8[2], o8[3]);
  *(float4*)(yrow + 4) = make_float4(o8[4], o8[5], o8[6], o8[7]);
}

// ---------------------------------------------------------------------------
// WMMA node transform: Y[r, ycol..+15] = relu?(X[r,:K] @ W[:, wcol..+15] + b)
// K templated (16 zero-padded to 32, or 32). One wave per 16-row tile.
// A: b128 global loads; B: LDS-staged weights; tail tile clamps row base
// (recomputes identical rows -> branch-free stores, EXEC all-ones for WMMA).
template <int K, int RELU>
__global__ __launch_bounds__(256) void wmma_xform_kernel(
    const float* __restrict__ X, int ldx,
    const float* __restrict__ W, int ldw, int wcol,
    const float* __restrict__ bias,
    float* __restrict__ Y, int ldy, int ycol,
    int tiles, int nrows) {
  __shared__ float sW[32 * 16];
  for (int t = threadIdx.x; t < 32 * 16; t += blockDim.x) {
    int k = t >> 4, cc = t & 15;
    sW[t] = (k < K) ? W[k * ldw + wcol + cc] : 0.f;
  }
  __syncthreads();

  int wave = (int)((blockIdx.x * blockDim.x + threadIdx.x) >> 5);
  if (wave >= tiles) return;  // wave-uniform
  int lane = threadIdx.x & 31;
  int half = lane >> 4;
  int l15 = lane & 15;
  int row0 = wave * 16;
  if (row0 + 16 > nrows) row0 = nrows - 16;  // tail clamp (nrows >= 16)

  // A fragment (ISA 7.12.2): lane holds row l15, K-halves split by lane group
  const float* xrow = X + (row0 + l15) * ldx;
  int kbA = half * 8;
  float4 p0 = ld4(xrow + kbA);
  float4 p1 = ld4(xrow + kbA + 4);
  v16h a;
  a[0] = (_Float16)p0.x; a[1] = (_Float16)p0.y;
  a[2] = (_Float16)p0.z; a[3] = (_Float16)p0.w;
  a[4] = (_Float16)p1.x; a[5] = (_Float16)p1.y;
  a[6] = (_Float16)p1.z; a[7] = (_Float16)p1.w;
  if (K == 32) {
    float4 p2 = ld4(xrow + 16 + kbA);
    float4 p3 = ld4(xrow + 16 + kbA + 4);
    a[8]  = (_Float16)p2.x; a[9]  = (_Float16)p2.y;
    a[10] = (_Float16)p2.z; a[11] = (_Float16)p2.w;
    a[12] = (_Float16)p3.x; a[13] = (_Float16)p3.y;
    a[14] = (_Float16)p3.z; a[15] = (_Float16)p3.w;
  } else {
#pragma unroll
    for (int i = 8; i < 16; ++i) a[i] = (_Float16)0.0f;
  }

  // B fragment: lane holds col l15, K 8h..8h+7 then 16+8h..16+8h+7 (from LDS)
  int kbB = half * 8;
  v16h b;
#pragma unroll
  for (int i = 0; i < 16; ++i) b[i] = (_Float16)sW[(kbB + i + (i >= 8 ? 8 : 0)) * 16 + l15];

  v8f c = {};
  c = __builtin_amdgcn_wmma_f32_16x16x32_f16(
      false, a, false, b, (short)0, c, false, false);

  float bv = bias[wcol + l15];
  int rbase = row0 + half * 8;
#pragma unroll
  for (int i = 0; i < 8; ++i) {
    float v = c[i] + bv;
    if (RELU) v = fmaxf(v, 0.f);
    Y[(rbase + i) * ldy + ycol + l15] = v;
  }
}

// ---------------------------------------------------------------------------
// GATv2 edge passes (H=4, C=4). rev selects edge direction.
__global__ void gat_logit_kernel(
    const float* __restrict__ xl, const float* __restrict__ xr,
    const float* __restrict__ ew, const int* __restrict__ ei, int E_, int rev,
    const float* __restrict__ We, const float* __restrict__ att,
    float* __restrict__ logit, float* __restrict__ maxb) {
  int e = blockIdx.x * blockDim.x + threadIdx.x;
  if (e >= E_) return;
  int s = ei[rev ? E_ + e : e];
  int d = ei[rev ? e : E_ + e];
  float w = ew[e];
  float lx[16], rx[16];
#pragma unroll
  for (int q = 0; q < 4; ++q) {
    float4 a4 = ld4(xl + s * 16 + q * 4);
    float4 b4 = ld4(xr + d * 16 + q * 4);
    lx[q * 4 + 0] = a4.x; lx[q * 4 + 1] = a4.y;
    lx[q * 4 + 2] = a4.z; lx[q * 4 + 3] = a4.w;
    rx[q * 4 + 0] = b4.x; rx[q * 4 + 1] = b4.y;
    rx[q * 4 + 2] = b4.z; rx[q * 4 + 3] = b4.w;
  }
  float lg[4];
#pragma unroll
  for (int h = 0; h < 4; ++h) {
    float acc = 0.f;
#pragma unroll
    for (int c = 0; c < 4; ++c) {
      int cc = h * 4 + c;
      float z = lx[cc] + rx[cc] + w * We[cc];
      z = (z > 0.f) ? z : 0.2f * z;  // leaky_relu 0.2
      acc += z * att[cc];
    }
    lg[h] = acc;
    atomicMaxF(&maxb[d * 4 + h], acc);
  }
  *(float4*)(logit + e * 4) = make_float4(lg[0], lg[1], lg[2], lg[3]);
}

__global__ void gat_expsum_kernel(
    const int* __restrict__ ei, int E_, int rev,
    float* __restrict__ logit, const float* __restrict__ maxb,
    float* __restrict__ sumb) {
  int e = blockIdx.x * blockDim.x + threadIdx.x;
  if (e >= E_) return;
  int d = ei[rev ? e : E_ + e];
  float4 lg = ld4(logit + e * 4);
  float4 mb = ld4(maxb + d * 4);
  float l[4] = {lg.x, lg.y, lg.z, lg.w};
  float m[4] = {mb.x, mb.y, mb.z, mb.w};
  float ex[4];
#pragma unroll
  for (int h = 0; h < 4; ++h) {
    float mm = (m[h] > -3.0e38f) ? m[h] : 0.f;
    ex[h] = __expf(l[h] - mm);
    atomicAdd(&sumb[d * 4 + h], ex[h]);
  }
  *(float4*)(logit + e * 4) = make_float4(ex[0], ex[1], ex[2], ex[3]);
}

__global__ void gat_scatter_kernel(
    const float* __restrict__ xl, const int* __restrict__ ei, int E_, int rev,
    const float* __restrict__ logit, const float* __restrict__ sumb,
    float* __restrict__ acc) {
  int e = blockIdx.x * blockDim.x + threadIdx.x;
  if (e >= E_) return;
  int s = ei[rev ? E_ + e : e];
  int d = ei[rev ? e : E_ + e];
  float4 lg = ld4(logit + e * 4);
  float4 sb = ld4(sumb + d * 4);
  float alpha[4] = {lg.x / (sb.x + 1e-16f), lg.y / (sb.y + 1e-16f),
                    lg.z / (sb.z + 1e-16f), lg.w / (sb.w + 1e-16f)};
#pragma unroll
  for (int q = 0; q < 4; ++q) {
    float4 v = ld4(xl + s * 16 + q * 4);
    float al = alpha[q];
    atomicAdd(&acc[d * 16 + q * 4 + 0], v.x * al);
    atomicAdd(&acc[d * 16 + q * 4 + 1], v.y * al);
    atomicAdd(&acc[d * 16 + q * 4 + 2], v.z * al);
    atomicAdd(&acc[d * 16 + q * 4 + 3], v.w * al);
  }
}

__global__ void combine_kernel(
    const float* __restrict__ accA, const float* __restrict__ biasA,
    const float* __restrict__ accB, const float* __restrict__ biasB,
    float* __restrict__ out, int N_) {
  int i = blockIdx.x * blockDim.x + threadIdx.x;
  if (i >= N_) return;
#pragma unroll
  for (int q = 0; q < 4; ++q) {
    float4 a4 = ld4(accA + i * 16 + q * 4);
    float4 b4 = ld4(accB + i * 16 + q * 4);
    float4 ba = ld4(biasA + q * 4);
    float4 bb = ld4(biasB + q * 4);
    float4 oa = make_float4(fmaxf(a4.x + ba.x, 0.f), fmaxf(a4.y + ba.y, 0.f),
                            fmaxf(a4.z + ba.z, 0.f), fmaxf(a4.w + ba.w, 0.f));
    float4 ob = make_float4(fmaxf(b4.x + bb.x, 0.f), fmaxf(b4.y + bb.y, 0.f),
                            fmaxf(b4.z + bb.z, 0.f), fmaxf(b4.w + bb.w, 0.f));
    *(float4*)(out + i * 32 + q * 4) = oa;
    *(float4*)(out + i * 32 + 16 + q * 4) = ob;
  }
}

// ---------------------------------------------------------------------------
// Final edge scores + per-src segment softmax over 2 columns.
__global__ void score_kernel(
    const float* __restrict__ hf, const float* __restrict__ ew,
    const int* __restrict__ ei, int E_,
    const float* __restrict__ W, const float* __restrict__ b,
    float* __restrict__ out, float* __restrict__ fmax) {
  int e = blockIdx.x * blockDim.x + threadIdx.x;
  if (e >= E_) return;
  int s = ei[e], d = ei[E_ + e];
  float w = ew[e];
  float a0 = b[0] + w * W[0];
  float a1 = b[1] + w * W[1];
#pragma unroll
  for (int q = 0; q < 8; ++q) {
    float4 hs = ld4(hf + s * 32 + q * 4);
    float hv[4] = {hs.x, hs.y, hs.z, hs.w};
#pragma unroll
    for (int t = 0; t < 4; ++t) {
      int c = q * 4 + t;
      a0 += hv[t] * W[(1 + c) * 2 + 0];
      a1 += hv[t] * W[(1 + c) * 2 + 1];
    }
  }
#pragma unroll
  for (int q = 0; q < 8; ++q) {
    float4 hd = ld4(hf + d * 32 + q * 4);
    float hv[4] = {hd.x, hd.y, hd.z, hd.w};
#pragma unroll
    for (int t = 0; t < 4; ++t) {
      int c = q * 4 + t;
      a0 += hv[t] * W[(33 + c) * 2 + 0];
      a1 += hv[t] * W[(33 + c) * 2 + 1];
    }
  }
  *(float2*)(out + e * 2) = make_float2(a0, a1);
  atomicMaxF(&fmax[s * 2 + 0], a0);
  atomicMaxF(&fmax[s * 2 + 1], a1);
}

__global__ void score_exp_kernel(
    const int* __restrict__ ei, int E_, const float* __restrict__ fmax,
    float* __restrict__ out, float* __restrict__ fsum) {
  int e = blockIdx.x * blockDim.x + threadIdx.x;
  if (e >= E_) return;
  int s = ei[e];
  float2 sc = *(const float2*)(out + e * 2);
  float2 mb = *(const float2*)(fmax + s * 2);
  float m0 = (mb.x > -3.0e38f) ? mb.x : 0.f;
  float m1 = (mb.y > -3.0e38f) ? mb.y : 0.f;
  float e0 = __expf(sc.x - m0), e1 = __expf(sc.y - m1);
  *(float2*)(out + e * 2) = make_float2(e0, e1);
  atomicAdd(&fsum[s * 2 + 0], e0);
  atomicAdd(&fsum[s * 2 + 1], e1);
}

__global__ void score_norm_kernel(
    const int* __restrict__ ei, int E_, const float* __restrict__ fsum,
    float* __restrict__ out) {
  int e = blockIdx.x * blockDim.x + threadIdx.x;
  if (e >= E_) return;
  int s = ei[e];
  float2 sc = *(const float2*)(out + e * 2);
  float2 sm = *(const float2*)(fsum + s * 2);
  *(float2*)(out + e * 2) =
      make_float2(sc.x / (sm.x + 1e-16f), sc.y / (sm.y + 1e-16f));
}

// ===========================================================================
extern "C" void kernel_launch(void* const* d_in, const int* in_sizes, int n_in,
                              void* d_out, int out_size, void* d_ws,
                              size_t ws_size, hipStream_t stream) {
  (void)n_in; (void)out_size; (void)ws_size;
  const float* x = (const float*)d_in[0];
  const float* ew = (const float*)d_in[1];
  int p = 2;
  const float* fceW = (const float*)d_in[p++];  // (65,2)
  const float* fceB = (const float*)d_in[p++];  // (2)
  const float* fcnW = (const float*)d_in[p++];  // (32,32)
  const float* fcnB = (const float*)d_in[p++];  // (32)
  struct GatP { const float *We, *Wl, *Wr, *att, *bias, *bl, *br; };
  struct PnaP { const float *We, *Wlin, *Wpost, *Wpre, *be, *blin, *bpost, *bpre; };
  GatP g1, g1c, g2;
  GatP* gs[3] = {&g1, &g1c, &g2};  // gat1, gat1_ce, gat2
  for (int i = 0; i < 3; ++i) {    // sorted keys: We, Wl, Wr, att, bias, bl, br
    GatP& g = *gs[i];
    g.We = (const float*)d_in[p++];   g.Wl = (const float*)d_in[p++];
    g.Wr = (const float*)d_in[p++];   g.att = (const float*)d_in[p++];
    g.bias = (const float*)d_in[p++]; g.bl = (const float*)d_in[p++];
    g.br = (const float*)d_in[p++];
  }
  PnaP pn0, pn1;
  PnaP* ps[2] = {&pn0, &pn1};      // pna, pna_ce
  for (int i = 0; i < 2; ++i) {    // sorted: We, Wlin, Wpost, Wpre, be, blin, bpost, bpre
    PnaP& q = *ps[i];
    q.We = (const float*)d_in[p++];    q.Wlin = (const float*)d_in[p++];
    q.Wpost = (const float*)d_in[p++]; q.Wpre = (const float*)d_in[p++];
    q.be = (const float*)d_in[p++];    q.blin = (const float*)d_in[p++];
    q.bpost = (const float*)d_in[p++]; q.bpre = (const float*)d_in[p++];
  }
  const int* ei = (const int*)d_in[p++];
  const int* pdeg = (const int*)d_in[p];
  const int BINS = in_sizes[p];

  const int N_ = in_sizes[0] / 5;
  const int E_ = in_sizes[1];

  // --- workspace layout (floats); all sub-buffers 16B-aligned
  float* ws = (float*)d_ws;
  size_t o = 0;
  float* avg = ws + o;   o += 64;
  float* h1 = ws + o;    o += (size_t)16 * N_;
  float* h2 = ws + o;    o += (size_t)32 * N_;
  float* h3 = ws + o;    o += (size_t)32 * N_;
  float* hf = ws + o;    o += (size_t)32 * N_;
  float* logit = ws + o; o += (size_t)4 * E_;
  float* pool = ws + o;  // reused: max(42N pna, 112N gat)
  float* xlA = pool;                    float* xrA = xlA + (size_t)16 * N_;
  float* xlB = xrA + (size_t)16 * N_;   float* xrB = xlB + (size_t)16 * N_;
  float* accA = xrB + (size_t)16 * N_;  float* accB = accA + (size_t)16 * N_;
  float* maxA = accB + (size_t)16 * N_; float* sumA = maxA + (size_t)4 * N_;
  float* maxB = sumA + (size_t)4 * N_;  float* sumB = maxB + (size_t)4 * N_;
  // PNA-phase overlay of pool
  float* cnt0 = pool;                 float* sum0 = cnt0 + N_;
  float* sq0 = sum0 + (size_t)5 * N_; float* mn0 = sq0 + (size_t)5 * N_;
  float* mx0 = mn0 + (size_t)5 * N_;
  float* cnt1 = mx0 + (size_t)5 * N_; float* sum1 = cnt1 + N_;
  float* sq1 = sum1 + (size_t)5 * N_; float* mn1 = sq1 + (size_t)5 * N_;
  float* mx1 = mn1 + (size_t)5 * N_;
  // final-phase overlay
  float* fmaxb = maxA;
  float* fsumb = sumA;

  const int TB = 256;
  const int ebl = (E_ + TB - 1) / TB;
  const int nbl = (N_ + TB - 1) / TB;
  const int tiles = (N_ + 15) / 16;
  const int wbl = (tiles + 7) / 8;  // 8 waves per 256-thread block
  const float INF = __builtin_huge_valf();
  float* dout = (float*)d_out;

  // ---- phase 0: degree scaler
  avg_log_kernel<<<1, 64, 0, stream>>>(pdeg, BINS, avg);

  // ---- phase 1: PNA (both directions)
  hipMemsetAsync(pool, 0, (size_t)42 * N_ * sizeof(float), stream);
  {
    int m5 = 5 * N_;
    int mbl = (m5 + TB - 1) / TB;
    fill_f32<<<mbl, TB, 0, stream>>>(mn0, INF, m5);
    fill_f32<<<mbl, TB, 0, stream>>>(mx0, -INF, m5);
    fill_f32<<<mbl, TB, 0, stream>>>(mn1, INF, m5);
    fill_f32<<<mbl, TB, 0, stream>>>(mx1, -INF, m5);
  }
  pna_edge_kernel<<<ebl, TB, 0, stream>>>(
      x, ew, ei, E_, pn0.We, pn0.be, pn0.Wpre, pn0.bpre,
      pn1.We, pn1.be, pn1.Wpre, pn1.bpre,
      cnt0, sum0, sq0, mn0, mx0, cnt1, sum1, sq1, mn1, mx1);
  pna_node_kernel<<<nbl, TB, 0, stream>>>(x, N_, cnt0, sum0, sq0, mn0, mx0,
      pn0.Wpost, pn0.bpost, pn0.Wlin, pn0.blin, avg, h1, 16, 0);
  pna_node_kernel<<<nbl, TB, 0, stream>>>(x, N_, cnt1, sum1, sq1, mn1, mx1,
      pn1.Wpost, pn1.bpost, pn1.Wlin, pn1.blin, avg, h1, 16, 8);

  // ---- phase 2: GAT1 (gat1 fwd + gat1_ce rev)
  hipMemsetAsync(accA, 0, (size_t)48 * N_ * sizeof(float), stream);
  {
    int m4 = 4 * N_;
    int hbl = (m4 + TB - 1) / TB;
    fill_f32<<<hbl, TB, 0, stream>>>(maxA, -INF, m4);
    fill_f32<<<hbl, TB, 0, stream>>>(maxB, -INF, m4);
  }
  wmma_xform_kernel<16, 0><<<wbl, TB, 0, stream>>>(h1, 16, g1.Wl, 16, 0, g1.bl, xlA, 16, 0, tiles, N_);
  wmma_xform_kernel<16, 0><<<wbl, TB, 0, stream>>>(h1, 16, g1.Wr, 16, 0, g1.br, xrA, 16, 0, tiles, N_);
  wmma_xform_kernel<16, 0><<<wbl, TB, 0, stream>>>(h1, 16, g1c.Wl, 16, 0, g1c.bl, xlB, 16, 0, tiles, N_);
  wmma_xform_kernel<16, 0><<<wbl, TB, 0, stream>>>(h1, 16, g1c.Wr, 16, 0, g1c.br, xrB, 16, 0, tiles, N_);

  gat_logit_kernel<<<ebl, TB, 0, stream>>>(xlA, xrA, ew, ei, E_, 0, g1.We, g1.att, logit, maxA);
  gat_expsum_kernel<<<ebl, TB, 0, stream>>>(ei, E_, 0, logit, maxA, sumA);
  gat_scatter_kernel<<<ebl, TB, 0, stream>>>(xlA, ei, E_, 0, logit, sumA, accA);

  gat_logit_kernel<<<ebl, TB, 0, stream>>>(xlB, xrB, ew, ei, E_, 1, g1c.We, g1c.att, logit, maxB);
  gat_expsum_kernel<<<ebl, TB, 0, stream>>>(ei, E_, 1, logit, maxB, sumB);
  gat_scatter_kernel<<<ebl, TB, 0, stream>>>(xlB, ei, E_, 1, logit, sumB, accB);

  combine_kernel<<<nbl, TB, 0, stream>>>(accA, g1.bias, accB, g1c.bias, h2, N_);

  // ---- phase 3: GAT2 (shared params, both directions; xl/xr computed once)
  hipMemsetAsync(accA, 0, (size_t)48 * N_ * sizeof(float), stream);
  {
    int m4 = 4 * N_;
    int hbl = (m4 + TB - 1) / TB;
    fill_f32<<<hbl, TB, 0, stream>>>(maxA, -INF, m4);
    fill_f32<<<hbl, TB, 0, stream>>>(maxB, -INF, m4);
  }
  wmma_xform_kernel<32, 0><<<wbl, TB, 0, stream>>>(h2, 32, g2.Wl, 16, 0, g2.bl, xlA, 16, 0, tiles, N_);
  wmma_xform_kernel<32, 0><<<wbl, TB, 0, stream>>>(h2, 32, g2.Wr, 16, 0, g2.br, xrA, 16, 0, tiles, N_);

  gat_logit_kernel<<<ebl, TB, 0, stream>>>(xlA, xrA, ew, ei, E_, 0, g2.We, g2.att, logit, maxA);
  gat_expsum_kernel<<<ebl, TB, 0, stream>>>(ei, E_, 0, logit, maxA, sumA);
  gat_scatter_kernel<<<ebl, TB, 0, stream>>>(xlA, ei, E_, 0, logit, sumA, accA);

  gat_logit_kernel<<<ebl, TB, 0, stream>>>(xlA, xrA, ew, ei, E_, 1, g2.We, g2.att, logit, maxB);
  gat_expsum_kernel<<<ebl, TB, 0, stream>>>(ei, E_, 1, logit, maxB, sumB);
  gat_scatter_kernel<<<ebl, TB, 0, stream>>>(xlA, ei, E_, 1, logit, sumB, accB);

  combine_kernel<<<nbl, TB, 0, stream>>>(accA, g2.bias, accB, g2.bias, h3, N_);

  // ---- phase 4: FCN (N x 32 x 32) + relu, two 16-col WMMA halves
  wmma_xform_kernel<32, 1><<<wbl, TB, 0, stream>>>(h3, 32, fcnW, 32, 0, fcnB, hf, 32, 0, tiles, N_);
  wmma_xform_kernel<32, 1><<<wbl, TB, 0, stream>>>(h3, 32, fcnW, 32, 16, fcnB, hf, 32, 16, tiles, N_);

  // ---- phase 5: edge scores + per-src segment softmax
  {
    int m2 = 2 * N_;
    int fbl = (m2 + TB - 1) / TB;
    fill_f32<<<fbl, TB, 0, stream>>>(fmaxb, -INF, m2);
    hipMemsetAsync(fsumb, 0, (size_t)2 * N_ * sizeof(float), stream);
  }
  score_kernel<<<ebl, TB, 0, stream>>>(hf, ew, ei, E_, fceW, fceB, dout, fmaxb);
  score_exp_kernel<<<ebl, TB, 0, stream>>>(ei, E_, fmaxb, dout, fsumb);
  score_norm_kernel<<<ebl, TB, 0, stream>>>(ei, E_, fsumb, dout);
}
